// GRUCell_59665685676320
// MI455X (gfx1250) — matrix operand order repaired and checked by reading, not compiled, not run on hardware
//
#include <hip/hip_runtime.h>
#include <math.h>

typedef __attribute__((ext_vector_type(16))) __bf16 v16bf;
typedef __attribute__((ext_vector_type(8)))  __bf16 v8bf;
typedef __attribute__((ext_vector_type(4)))  __bf16 v4bf;
typedef __attribute__((ext_vector_type(8)))  float  v8f;

#define BN_EPS 1e-5f

static constexpr int Bdim = 4096;   // batch (M)
static constexpr int Kdim = 1024;   // I == H == K of both GEMMs
static constexpr int Hdim = 1024;   // hidden
static constexpr int N3   = 3072;   // 3*H

// ---------------- helpers ----------------

__device__ __forceinline__ void bf16_split(float x, __bf16& h, __bf16& l) {
    h = (__bf16)x;                  // hi ~ top 8 mantissa bits
    l = (__bf16)(x - (float)h);     // lo ~ next 8 bits
}

__device__ __forceinline__ v8f wmma_bf16(v16bf a, v16bf b, v8f c) {
    return __builtin_amdgcn_wmma_f32_16x16x32_bf16(
        false, a, false, b, (short)0, c, false, false);
}

__device__ __forceinline__ v16bf cat8(v8bf a, v8bf b) {
    return __builtin_shufflevector(a, b, 0,1,2,3,4,5,6,7,8,9,10,11,12,13,14,15);
}

__device__ __forceinline__ float sigmoid_f(float x) {
    return 1.0f / (1.0f + __expf(-x));
}

// ---------------- kernel 0a: elementwise fp32 -> bf16 hi/lo split ----------------
// one float4 per thread; grid = count/1024

__global__ __launch_bounds__(256) void convert_split_kernel(
    const float* __restrict__ src, __bf16* __restrict__ hi, __bf16* __restrict__ lo)
{
    const size_t i = (size_t)(blockIdx.x * 256 + threadIdx.x) * 4;
    float4 v = *(const float4*)(src + i);
    float xs[4] = {v.x, v.y, v.z, v.w};
    v4bf h, l;
    #pragma unroll
    for (int e = 0; e < 4; ++e) {
        __bf16 hh, ll; bf16_split(xs[e], hh, ll);
        h[e] = hh; l[e] = ll;
    }
    *(v4bf*)(hi + i) = h;
    *(v4bf*)(lo + i) = l;
}

// ---------------- kernel 0b: weight [K][N] -> transposed bf16 hi/lo [N][K] ----------------
// block (32,8), 32x32 tiles via LDS; grid (N/32, K/32)

__global__ __launch_bounds__(256) void convert_transpose_kernel(
    const float* __restrict__ W,             // [Kdim][N3]
    __bf16* __restrict__ Thi, __bf16* __restrict__ Tlo)  // [N3][Kdim]
{
    __shared__ __bf16 Lhi[32][33];
    __shared__ __bf16 Llo[32][33];
    const int n0 = blockIdx.x * 32, k0 = blockIdx.y * 32;
    const int tx = threadIdx.x, ty = threadIdx.y;
    #pragma unroll
    for (int i = 0; i < 4; ++i) {
        const int k = k0 + ty + i * 8;
        float x = W[(size_t)k * N3 + n0 + tx];
        __bf16 h, l; bf16_split(x, h, l);
        Lhi[ty + i * 8][tx] = h;
        Llo[ty + i * 8][tx] = l;
    }
    __syncthreads();
    #pragma unroll
    for (int i = 0; i < 4; ++i) {
        const int n = n0 + ty + i * 8;
        Thi[(size_t)n * Kdim + k0 + tx] = Lhi[tx][ty + i * 8];
        Tlo[(size_t)n * Kdim + k0 + tx] = Llo[tx][ty + i * 8];
    }
}

// ---------------- kernel 1: G = input @ weight_i (bf16x3, pre-split operands) ----------------
// 128x128 block tile, 256 threads = 8 waves (4 m x 2 n), wave tile 32x64

__global__ __launch_bounds__(256) void gemm_pre_kernel(
    const __bf16* __restrict__ gAhi, const __bf16* __restrict__ gAlo,  // [Bdim][Kdim]
    const __bf16* __restrict__ gBhi, const __bf16* __restrict__ gBlo,  // [N3][Kdim] (transposed)
    float* __restrict__ G)                                             // [Bdim][N3]
{
    __shared__ __align__(32) __bf16 sAhi[128 * 32];
    __shared__ __align__(32) __bf16 sAlo[128 * 32];
    __shared__ __align__(32) __bf16 sBhi[128 * 32];   // [n][k]
    __shared__ __align__(32) __bf16 sBlo[128 * 32];

    const int tid   = threadIdx.x;
    const int mBase = blockIdx.y * 128;
    const int nBase = blockIdx.x * 128;
    const int wave  = tid >> 5;
    const int lane  = tid & 31;
    const int w0    = wave & 1;       // n half (64 each)
    const int w1    = wave >> 1;      // m quarter (32 each)
    const int lm    = lane & 15;
    const int lh    = lane >> 4;

    v8f acc[2][4] = {};

    for (int kc = 0; kc < Kdim; kc += 32) {
        // ---- stage 4 x (128 rows x 32 k) bf16 tiles: pure b128 copies ----
        #pragma unroll
        for (int i = 0; i < 2; ++i) {
            const int p = i * 256 + tid;        // 0..511
            const int row = p >> 2, seg = p & 3;
            *(v8bf*)(sAhi + row * 32 + seg * 8) =
                *(const v8bf*)(gAhi + (size_t)(mBase + row) * Kdim + kc + seg * 8);
            *(v8bf*)(sAlo + row * 32 + seg * 8) =
                *(const v8bf*)(gAlo + (size_t)(mBase + row) * Kdim + kc + seg * 8);
            *(v8bf*)(sBhi + row * 32 + seg * 8) =
                *(const v8bf*)(gBhi + (size_t)(nBase + row) * Kdim + kc + seg * 8);
            *(v8bf*)(sBlo + row * 32 + seg * 8) =
                *(const v8bf*)(gBlo + (size_t)(nBase + row) * Kdim + kc + seg * 8);
        }
        if (kc + 32 < Kdim) {  // prefetch next chunk (global_prefetch_b8)
            const int p = tid, row = p >> 2, seg = p & 3;
            __builtin_prefetch(gAhi + (size_t)(mBase + row) * Kdim + kc + 32 + seg * 8, 0, 1);
            __builtin_prefetch(gBhi + (size_t)(nBase + row) * Kdim + kc + 32 + seg * 8, 0, 1);
        }
        __syncthreads();

        #pragma unroll
        for (int mi = 0; mi < 2; ++mi) {
            const int arow = w1 * 32 + mi * 16 + lm;
            const int ak   = lh * 16;
            v16bf Ah = cat8(*(const v8bf*)(sAhi + arow * 32 + ak),
                            *(const v8bf*)(sAhi + arow * 32 + ak + 8));
            v16bf Al = cat8(*(const v8bf*)(sAlo + arow * 32 + ak),
                            *(const v8bf*)(sAlo + arow * 32 + ak + 8));
            #pragma unroll
            for (int ni = 0; ni < 4; ++ni) {
                const int bn = w0 * 64 + ni * 16 + lm;
                const int bk = lh * 16;
                v16bf Bh = *(const v16bf*)(sBhi + bn * 32 + bk);
                v16bf Bl = *(const v16bf*)(sBlo + bn * 32 + bk);
                acc[mi][ni] = wmma_bf16(Ah, Bh, acc[mi][ni]);
                acc[mi][ni] = wmma_bf16(Ah, Bl, acc[mi][ni]);
                acc[mi][ni] = wmma_bf16(Al, Bh, acc[mi][ni]);
            }
        }
        __syncthreads();
    }

    #pragma unroll
    for (int mi = 0; mi < 2; ++mi) {
        #pragma unroll
        for (int ni = 0; ni < 4; ++ni) {
            const int col = nBase + w0 * 64 + ni * 16 + lm;
            #pragma unroll
            for (int v = 0; v < 8; ++v) {
                const int row = mBase + w1 * 32 + mi * 16 + lh * 8 + v;
                G[(size_t)row * N3 + col] = acc[mi][ni][v];
            }
        }
    }
}

// ---------------- kernel 2a/2b: deterministic two-pass BN stats ----------------

__global__ __launch_bounds__(256) void bn_partial_kernel(
    const float* __restrict__ G, float* __restrict__ psum, float* __restrict__ psq)
{
    const int j  = blockIdx.x * 256 + threadIdx.x;
    const int r0 = blockIdx.y * (Bdim / 16);
    float s = 0.0f, q = 0.0f;
    for (int r = 0; r < Bdim / 16; ++r) {
        float x = G[(size_t)(r0 + r) * N3 + j];
        s += x; q += x * x;
    }
    psum[blockIdx.y * N3 + j] = s;
    psq [blockIdx.y * N3 + j] = q;
}

__global__ __launch_bounds__(256) void bn_finalize_kernel(
    const float* __restrict__ psum, const float* __restrict__ psq,
    float* __restrict__ mean, float* __restrict__ var)
{
    const int j = blockIdx.x * 256 + threadIdx.x;
    float s = 0.0f, q = 0.0f;
    #pragma unroll
    for (int i = 0; i < 16; ++i) { s += psum[i * N3 + j]; q += psq[i * N3 + j]; }
    const float m = s * (1.0f / Bdim);
    mean[j] = m;
    var[j]  = q * (1.0f / Bdim) - m * m;
}

// ---------------- kernel 3: fused hx@weight_h (3 gates) + BN + GRU gating ----------------
// 128x64 block tile, 256 threads = 8 waves (4 m x 2 n), wave tile 32x32, 3 gate accumulators

__global__ __launch_bounds__(256) void gru_fused_kernel(
    const __bf16* __restrict__ gAhi, const __bf16* __restrict__ gAlo,  // hx split [Bdim][Kdim]
    const __bf16* __restrict__ gBhi, const __bf16* __restrict__ gBlo,  // weight_h^T split [N3][Kdim]
    const float* __restrict__ hx,
    const float* __restrict__ G,
    const float* __restrict__ mean, const float* __restrict__ var,
    const float* __restrict__ gamma, const float* __restrict__ beta,
    const float* __restrict__ bias,
    float* __restrict__ out)          // [Bdim][Hdim]
{
    __shared__ __align__(32) __bf16 sAhi[128 * 32];
    __shared__ __align__(32) __bf16 sAlo[128 * 32];
    __shared__ __align__(32) __bf16 sBhi[3][64 * 32];
    __shared__ __align__(32) __bf16 sBlo[3][64 * 32];

    const int tid   = threadIdx.x;
    const int mBase = blockIdx.y * 128;
    const int nBase = blockIdx.x * 64;       // within [0,Hdim)
    const int wave  = tid >> 5;
    const int lane  = tid & 31;
    const int w0    = wave & 1;       // n half (32 each)
    const int w1    = wave >> 1;      // m quarter (32 each)
    const int lm    = lane & 15;
    const int lh    = lane >> 4;

    v8f acc[3][2][2] = {};            // [gate][mi][ci]

    for (int kc = 0; kc < Kdim; kc += 32) {
        // ---- stage A (128x32) ----
        #pragma unroll
        for (int i = 0; i < 2; ++i) {
            const int p = i * 256 + tid;
            const int row = p >> 2, seg = p & 3;
            *(v8bf*)(sAhi + row * 32 + seg * 8) =
                *(const v8bf*)(gAhi + (size_t)(mBase + row) * Kdim + kc + seg * 8);
            *(v8bf*)(sAlo + row * 32 + seg * 8) =
                *(const v8bf*)(gAlo + (size_t)(mBase + row) * Kdim + kc + seg * 8);
        }
        // ---- stage B per gate (64x32 each, rows nBase + g*Hdim in W^T) ----
        #pragma unroll
        for (int g = 0; g < 3; ++g) {
            const int p = tid;               // 256 slots exactly
            const int row = p >> 2, seg = p & 3;
            const size_t src = (size_t)(nBase + g * Hdim + row) * Kdim + kc + seg * 8;
            *(v8bf*)(sBhi[g] + row * 32 + seg * 8) = *(const v8bf*)(gBhi + src);
            *(v8bf*)(sBlo[g] + row * 32 + seg * 8) = *(const v8bf*)(gBlo + src);
        }
        if (kc + 32 < Kdim) {
            const int row = tid >> 2, seg = tid & 3;
            __builtin_prefetch(gAhi + (size_t)(mBase + row) * Kdim + kc + 32 + seg * 8, 0, 1);
        }
        __syncthreads();

        #pragma unroll
        for (int mi = 0; mi < 2; ++mi) {
            const int arow = w1 * 32 + mi * 16 + lm;
            const int ak   = lh * 16;
            v16bf Ah = cat8(*(const v8bf*)(sAhi + arow * 32 + ak),
                            *(const v8bf*)(sAhi + arow * 32 + ak + 8));
            v16bf Al = cat8(*(const v8bf*)(sAlo + arow * 32 + ak),
                            *(const v8bf*)(sAlo + arow * 32 + ak + 8));
            #pragma unroll
            for (int ci = 0; ci < 2; ++ci) {
                const int bn = w0 * 32 + ci * 16 + lm;
                const int bk = lh * 16;
                #pragma unroll
                for (int g = 0; g < 3; ++g) {
                    v16bf Bh = *(const v16bf*)(sBhi[g] + bn * 32 + bk);
                    v16bf Bl = *(const v16bf*)(sBlo[g] + bn * 32 + bk);
                    acc[g][mi][ci] = wmma_bf16(Ah, Bh, acc[g][mi][ci]);
                    acc[g][mi][ci] = wmma_bf16(Ah, Bl, acc[g][mi][ci]);
                    acc[g][mi][ci] = wmma_bf16(Al, Bh, acc[g][mi][ci]);
                }
            }
        }
        __syncthreads();
    }

    // ---- epilogue: BN + bias + activations + GRU gating ----
    #pragma unroll
    for (int mi = 0; mi < 2; ++mi) {
        #pragma unroll
        for (int ci = 0; ci < 2; ++ci) {
            const int j = nBase + w0 * 32 + ci * 16 + lm;
            const int jU = j, jR = j + Hdim, jC = j + 2 * Hdim;

            const float sU = rsqrtf(var[jU] + BN_EPS) * gamma[jU], mU = mean[jU];
            const float sR = rsqrtf(var[jR] + BN_EPS) * gamma[jR], mR = mean[jR];
            const float sC = rsqrtf(var[jC] + BN_EPS) * gamma[jC], mC = mean[jC];
            const float oU = beta[jU] + bias[jU];
            const float oR = beta[jR] + bias[jR];
            const float oC = beta[jC] + bias[jC];

            #pragma unroll
            for (int v = 0; v < 8; ++v) {
                const int b = mBase + w1 * 32 + mi * 16 + lh * 8 + v;
                const size_t gb = (size_t)b * N3;
                const float gnU = (G[gb + jU] - mU) * sU + oU;
                const float gnR = (G[gb + jR] - mR) * sR + oR;
                const float gnC = (G[gb + jC] - mC) * sC + oC;

                const float u  = sigmoid_f(gnU + acc[0][mi][ci][v]);
                const float rr = sigmoid_f(gnR + acc[1][mi][ci][v]);
                const float cc = tanhf(gnC + rr * acc[2][mi][ci][v]);
                const float hv = hx[(size_t)b * Hdim + j];
                out[(size_t)b * Hdim + j] = (1.0f - u) * hv + u * cc;
            }
        }
    }
}

// ---------------- launcher ----------------

extern "C" void kernel_launch(void* const* d_in, const int* in_sizes, int n_in,
                              void* d_out, int out_size, void* d_ws, size_t ws_size,
                              hipStream_t stream) {
    (void)in_sizes; (void)n_in; (void)out_size; (void)ws_size;

    const float* input = (const float*)d_in[0];
    const float* hx    = (const float*)d_in[1];
    const float* wi    = (const float*)d_in[2];
    const float* wh    = (const float*)d_in[3];
    const float* bias  = (const float*)d_in[4];
    const float* gamma = (const float*)d_in[5];
    const float* beta  = (const float*)d_in[6];
    float* out = (float*)d_out;

    // workspace layout
    char* ws = (char*)d_ws;
    const size_t szG   = (size_t)Bdim * N3 * sizeof(float);       // 50.3 MB
    const size_t szP   = (size_t)16 * N3 * sizeof(float);
    const size_t szS   = (size_t)N3 * sizeof(float);
    const size_t szAct = (size_t)Bdim * Kdim * sizeof(__bf16);    // 8.4 MB
    const size_t szWt  = (size_t)N3 * Kdim * sizeof(__bf16);      // 6.3 MB

    float*  G     = (float*)ws;                      ws += szG;
    float*  psum  = (float*)ws;                      ws += szP;
    float*  psq   = (float*)ws;                      ws += szP;
    float*  mean  = (float*)ws;                      ws += szS;
    float*  var   = (float*)ws;                      ws += szS;
    __bf16* Ahi   = (__bf16*)ws;                     ws += szAct;
    __bf16* Alo   = (__bf16*)ws;                     ws += szAct;
    __bf16* HXhi  = (__bf16*)ws;                     ws += szAct;
    __bf16* HXlo  = (__bf16*)ws;                     ws += szAct;
    __bf16* WIThi = (__bf16*)ws;                     ws += szWt;
    __bf16* WITlo = (__bf16*)ws;                     ws += szWt;
    __bf16* WHThi = (__bf16*)ws;                     ws += szWt;
    __bf16* WHTlo = (__bf16*)ws;                     ws += szWt;

    // 0) one-time precision split + weight transpose (~60 MB traffic, ~3 us)
    convert_split_kernel<<<(Bdim * Kdim) / 1024, 256, 0, stream>>>(input, Ahi, Alo);
    convert_split_kernel<<<(Bdim * Kdim) / 1024, 256, 0, stream>>>(hx, HXhi, HXlo);
    convert_transpose_kernel<<<dim3(N3 / 32, Kdim / 32), dim3(32, 8), 0, stream>>>(wi, WIThi, WITlo);
    convert_transpose_kernel<<<dim3(N3 / 32, Kdim / 32), dim3(32, 8), 0, stream>>>(wh, WHThi, WHTlo);

    // 1) G = input @ weight_i
    gemm_pre_kernel<<<dim3(N3 / 128, Bdim / 128), 256, 0, stream>>>(Ahi, Alo, WIThi, WITlo, G);

    // 2) BN statistics
    bn_partial_kernel<<<dim3(N3 / 256, 16), 256, 0, stream>>>(G, psum, psq);
    bn_finalize_kernel<<<dim3(N3 / 256), 256, 0, stream>>>(psum, psq, mean, var);

    // 3) fused hx@weight_h + BN + GRU gate
    gru_fused_kernel<<<dim3(Hdim / 64, Bdim / 128), 256, 0, stream>>>(
        HXhi, HXlo, WHThi, WHTlo, hx, G, mean, var, gamma, beta, bias, out);
}
